// SimpleAttention_28827820490760
// MI455X (gfx1250) — compile-verified, matrix operand order checked by reading, not run on hardware
//
#include <hip/hip_runtime.h>

typedef __attribute__((ext_vector_type(16))) __bf16 v16bf;
typedef __attribute__((ext_vector_type(8)))  __bf16 v8bf;
typedef __attribute__((ext_vector_type(8)))  float  v8f;
typedef __attribute__((ext_vector_type(4)))  float  v4f;
typedef __attribute__((ext_vector_type(4)))  int    v4i;
typedef __attribute__((ext_vector_type(4)))  unsigned short v4u16;

#define D_MODEL 512
#define SEQ     2048
#define BATCH   8

#if defined(__has_builtin)
#  if __has_builtin(__builtin_amdgcn_global_load_async_to_lds_b128)
#    define HAVE_ASYNC_LDS 1
#  endif
#endif

__device__ __forceinline__ unsigned short f32_to_bf16(float f) {
  union { float f; unsigned u; } x; x.f = f;
  unsigned r = (x.u + 0x7FFFu + ((x.u >> 16) & 1u)) >> 16;  // RNE
  return (unsigned short)r;
}

union BFrag { v16bf v; v8bf h[2]; };

// Load a 16x32 bf16 WMMA operand fragment from a row-major matrix (global or LDS).
// Lane l: row = l&15, K-half = l>>4; two 16B chunks at +kh*8 and +16+kh*8.
__device__ __forceinline__ BFrag load_frag(const unsigned short* __restrict__ base,
                                           int row0, int col0, int ld, int lane) {
  const int r = lane & 15, hh = lane >> 4;
  const unsigned short* p = base + (size_t)(row0 + r) * ld + col0 + hh * 8;
  BFrag f;
  f.h[0] = *(const v8bf*)p;
  f.h[1] = *(const v8bf*)(p + 16);
  return f;
}

// 16B async DMA global -> LDS (falls back to a synchronous copy through VGPRs).
__device__ __forceinline__ void async_b128(const unsigned short* g, unsigned short* l) {
#if defined(HAVE_ASYNC_LDS)
  typedef __attribute__((address_space(1))) v4i gv4i;   // global int4
  typedef __attribute__((address_space(3))) v4i lv4i;   // LDS int4
  __builtin_amdgcn_global_load_async_to_lds_b128(
      (gv4i*)(uintptr_t)g,
      (lv4i*)(unsigned)(uintptr_t)l,   // low 32 bits of flat shared addr = LDS offset
      0, 0);
#else
  *(v8bf*)l = *(const v8bf*)g;
#endif
}

#if defined(HAVE_ASYNC_LDS)
#  if __has_builtin(__builtin_amdgcn_s_wait_asynccnt)
#    define WAIT_ASYNC_LE2() __builtin_amdgcn_s_wait_asynccnt(2)
#    define WAIT_ASYNC_0()   __builtin_amdgcn_s_wait_asynccnt(0)
#  else
#    define WAIT_ASYNC_LE2() asm volatile("s_wait_asynccnt 0x2" ::: "memory")
#    define WAIT_ASYNC_0()   asm volatile("s_wait_asynccnt 0x0" ::: "memory")
#  endif
#else
#  define WAIT_ASYNC_LE2()
#  define WAIT_ASYNC_0()
#endif

// ---------------- x: f32 -> bf16 (row-major, 4 elems/thread) ----------------
__global__ __launch_bounds__(256) void k_cvt_x(const float* __restrict__ x,
                                               unsigned short* __restrict__ xb) {
  const int i = blockIdx.x * 256 + threadIdx.x;
  v4f v = ((const v4f*)x)[i];
  v4u16 o;
  o.x = f32_to_bf16(v.x); o.y = f32_to_bf16(v.y);
  o.z = f32_to_bf16(v.z); o.w = f32_to_bf16(v.w);
  ((v4u16*)xb)[i] = o;
}

// ---------------- W (D x E) f32 -> Wt (E x D) bf16, one z per weight --------
__global__ __launch_bounds__(256) void k_cvt_w(const float* __restrict__ Wq,
                                               const float* __restrict__ Wk,
                                               const float* __restrict__ Wv,
                                               unsigned short* __restrict__ Wt) {
  const float* src = (blockIdx.z == 0) ? Wq : (blockIdx.z == 1) ? Wk : Wv;
  unsigned short* dst = Wt + (size_t)blockIdx.z * D_MODEL * D_MODEL;
  const int i = blockIdx.x * 256 + threadIdx.x;
  const int e = i >> 9, d = i & 511;
  dst[i] = f32_to_bf16(src[d * D_MODEL + e]);
}

// ---------------- Projections: P = Xb @ Wt^T + bias, z in {Q,K,V} -----------
// Block tile 128(M) x 128(N). The shared 128x32 B-tile is DMA'd to LDS with
// async global->LDS loads, double-buffered against the WMMA stream.
__global__ __launch_bounds__(256) void k_proj(const unsigned short* __restrict__ Xb,
                                              const unsigned short* __restrict__ Wt,
                                              const float* __restrict__ bq,
                                              const float* __restrict__ bk,
                                              const float* __restrict__ bv,
                                              unsigned short* __restrict__ Qb,
                                              unsigned short* __restrict__ Kb,
                                              unsigned short* __restrict__ Vt) {
  __shared__ __align__(16) unsigned short btile[2][128][32];   // 2 x 8 KB

  const int tid  = threadIdx.x;
  const int lane = tid & 31;
  const int w    = tid >> 5;
  const int z    = blockIdx.z;
  const int m0   = blockIdx.x * 128 + w * 16;
  const int n0   = blockIdx.y * 128;
  const unsigned short* Wz = Wt + (size_t)z * D_MODEL * D_MODEL;
  const float* bias = (z == 0) ? bq : (z == 1) ? bk : bv;

  // Cooperative staging: thread -> (row, 16-elem chunk) of the 128x32 tile.
  const int srow = tid >> 1;
  const int sch  = (tid & 1) * 16;

  {
    const unsigned short* g = Wz + (size_t)(n0 + srow) * D_MODEL + sch;
    unsigned short* l = &btile[0][srow][sch];
    async_b128(g, l);
    async_b128(g + 8, l + 8);
  }

  const v8f z8 = {0.f,0.f,0.f,0.f,0.f,0.f,0.f,0.f};
  v8f c[8];
  #pragma unroll
  for (int t = 0; t < 8; ++t) c[t] = z8;

  for (int kk = 0; kk < 16; ++kk) {
    const int buf = kk & 1;
    if (kk + 1 < 16) {                       // prefetch next B tile via DMA
      const unsigned short* g = Wz + (size_t)(n0 + srow) * D_MODEL + (kk + 1) * 32 + sch;
      unsigned short* l = &btile[buf ^ 1][srow][sch];
      async_b128(g, l);
      async_b128(g + 8, l + 8);
      WAIT_ASYNC_LE2();                      // current tile's DMA done (in-order)
    } else {
      WAIT_ASYNC_0();
    }
    __syncthreads();                         // tile visible to all waves

    BFrag a = load_frag(Xb, m0, kk * 32, D_MODEL, lane);
    #pragma unroll
    for (int nt = 0; nt < 8; ++nt) {
      BFrag bfr = load_frag(&btile[buf][0][0], nt * 16, 0, 32, lane);
      c[nt] = __builtin_amdgcn_wmma_f32_16x16x32_bf16(false, a.v, false, bfr.v,
                                                      (short)0, c[nt], false, false);
    }
    __syncthreads();                         // reads done before next overwrite
  }

  const int r = lane & 15, hh = lane >> 4;
  #pragma unroll
  for (int nt = 0; nt < 8; ++nt) {
    const int e = n0 + nt * 16 + r;
    const float be = bias[e];
    #pragma unroll
    for (int g = 0; g < 8; ++g) {
      const int row = m0 + g + 8 * hh;
      const unsigned short val = f32_to_bf16(c[nt][g] + be);
      if (z == 0)      Qb[(size_t)row * D_MODEL + e] = val;
      else if (z == 1) Kb[(size_t)row * D_MODEL + e] = val;
      else {                                 // V stored transposed (b, e, s)
        const int bb = row >> 11, s = row & 2047;
        Vt[(size_t)bb * D_MODEL * SEQ + (size_t)e * SEQ + s] = val;
      }
    }
  }
}

// ---------------- Fused flash attention -------------------------------------
// Block = 16 query rows; 8 waves split D=512 into 64-wide slices.
__global__ __launch_bounds__(256) void k_attn(const unsigned short* __restrict__ Qb,
                                              const unsigned short* __restrict__ Kb,
                                              const unsigned short* __restrict__ Vt,
                                              float* __restrict__ out) {
  __shared__ __align__(32) float sred[8][32][16];        // per-wave partial scores
  __shared__ __align__(16) unsigned short pbuf[16][32];  // P tile, bf16 row-major

  const int lane = threadIdx.x & 31;
  const int w    = threadIdx.x >> 5;
  const int r    = lane & 15, hh = lane >> 4;
  const int q0   = blockIdx.x * 16;
  const int b    = blockIdx.y;
  const float scale = 0.04419417382415922f;              // 1/sqrt(512)

  const unsigned short* Qbase = Qb + (size_t)b * SEQ * D_MODEL;
  const unsigned short* Kbase = Kb + (size_t)b * SEQ * D_MODEL;
  const unsigned short* Vbase = Vt + (size_t)b * D_MODEL * SEQ;

  BFrag aq[2];
  #pragma unroll
  for (int ks = 0; ks < 2; ++ks)
    aq[ks] = load_frag(Qbase, q0, w * 64 + ks * 32, D_MODEL, lane);

  const v8f z8 = {0.f,0.f,0.f,0.f,0.f,0.f,0.f,0.f};
  v8f o[4];
  #pragma unroll
  for (int t = 0; t < 4; ++t) o[t] = z8;
  float m_run[8], l_run[8];
  #pragma unroll
  for (int g = 0; g < 8; ++g) { m_run[g] = -1e30f; l_run[g] = 0.f; }

  for (int kc = 0; kc < SEQ; kc += 32) {
    if (kc + 32 < SEQ) {                                 // cover next chunk's latency
      __builtin_prefetch(Kbase + (size_t)(kc + 32 + r) * D_MODEL + w * 64 + hh * 8, 0, 0);
      __builtin_prefetch(Vbase + (size_t)(w * 64 + r) * SEQ + kc + 32 + hh * 8, 0, 0);
    }

    // 1) partial scores over this wave's D slice: S_w = Q_w @ Kc_w^T (16x32)
    v8f s0 = z8, s1 = z8;
    #pragma unroll
    for (int ks = 0; ks < 2; ++ks) {
      const int d = w * 64 + ks * 32;
      BFrag bk0 = load_frag(Kbase, kc,      d, D_MODEL, lane);
      s0 = __builtin_amdgcn_wmma_f32_16x16x32_bf16(false, aq[ks].v, false, bk0.v,
                                                   (short)0, s0, false, false);
      BFrag bk1 = load_frag(Kbase, kc + 16, d, D_MODEL, lane);
      s1 = __builtin_amdgcn_wmma_f32_16x16x32_bf16(false, aq[ks].v, false, bk1.v,
                                                   (short)0, s1, false, false);
    }

    __syncthreads();                                     // prev iter done with sred/pbuf
    v8f* sp = (v8f*)&sred[w][lane][0];                   // store in native C layout
    sp[0] = s0; sp[1] = s1;
    __syncthreads();

    // 2) cross-wave reduction of the D-split partials (layout-preserving)
    v8f r0 = z8, r1 = z8;
    #pragma unroll
    for (int w2 = 0; w2 < 8; ++w2) {
      const v8f* p = (const v8f*)&sred[w2][lane][0];
      r0 += p[0]; r1 += p[1];
    }
    r0 *= scale; r1 *= scale;

    // 3) online softmax (redundant per wave; rows = vgpr + 8*lane-half)
    float alpha[8];
    #pragma unroll
    for (int g = 0; g < 8; ++g) {
      float mx = fmaxf(r0[g], r1[g]);
      #pragma unroll
      for (int sh = 8; sh >= 1; sh >>= 1) mx = fmaxf(mx, __shfl_xor(mx, sh, 32));
      const float mn = fmaxf(m_run[g], mx);
      alpha[g] = __expf(m_run[g] - mn);
      const float p0 = __expf(r0[g] - mn);
      const float p1 = __expf(r1[g] - mn);
      r0[g] = p0; r1[g] = p1;
      float rs = p0 + p1;
      #pragma unroll
      for (int sh = 8; sh >= 1; sh >>= 1) rs += __shfl_xor(rs, sh, 32);
      l_run[g] = l_run[g] * alpha[g] + rs;
      m_run[g] = mn;
    }

    // 4) publish P (bf16) once
    if (w == 0) {
      #pragma unroll
      for (int g = 0; g < 8; ++g) {
        pbuf[g + 8 * hh][r]      = f32_to_bf16(r0[g]);
        pbuf[g + 8 * hh][16 + r] = f32_to_bf16(r1[g]);
      }
    }
    __syncthreads();

    // 5) rescale O and accumulate O += P @ Vc over this wave's 64 output dims
    #pragma unroll
    for (int t = 0; t < 4; ++t)
      #pragma unroll
      for (int g = 0; g < 8; ++g) o[t][g] *= alpha[g];

    BFrag pf;
    pf.h[0] = *(const v8bf*)&pbuf[r][hh * 8];
    pf.h[1] = *(const v8bf*)&pbuf[r][16 + hh * 8];

    #pragma unroll
    for (int nt = 0; nt < 4; ++nt) {
      BFrag vf = load_frag(Vbase, w * 64 + nt * 16, kc, SEQ, lane);
      o[nt] = __builtin_amdgcn_wmma_f32_16x16x32_bf16(false, pf.v, false, vf.v,
                                                      (short)0, o[nt], false, false);
    }
  }

  #pragma unroll
  for (int nt = 0; nt < 4; ++nt) {
    #pragma unroll
    for (int g = 0; g < 8; ++g) {
      const int row = q0 + g + 8 * hh;
      out[((size_t)b * SEQ + row) * D_MODEL + w * 64 + nt * 16 + r] = o[nt][g] / l_run[g];
    }
  }
}

extern "C" void kernel_launch(void* const* d_in, const int* in_sizes, int n_in,
                              void* d_out, int out_size, void* d_ws, size_t ws_size,
                              hipStream_t stream) {
  (void)in_sizes; (void)n_in; (void)out_size; (void)ws_size;
  const float* x  = (const float*)d_in[0];
  const float* Wq = (const float*)d_in[1];
  const float* bq = (const float*)d_in[2];
  const float* Wk = (const float*)d_in[3];
  const float* bk = (const float*)d_in[4];
  const float* Wv = (const float*)d_in[5];
  const float* bv = (const float*)d_in[6];
  float* out = (float*)d_out;

  char* ws = (char*)d_ws;                                 // needs ~65.5 MB
  unsigned short* Xb = (unsigned short*)(ws + ((size_t)0  << 20));  // 16 MB
  unsigned short* Qb = (unsigned short*)(ws + ((size_t)16 << 20));  // 16 MB
  unsigned short* Kb = (unsigned short*)(ws + ((size_t)32 << 20));  // 16 MB
  unsigned short* Vt = (unsigned short*)(ws + ((size_t)48 << 20));  // 16 MB (transposed)
  unsigned short* Wt = (unsigned short*)(ws + ((size_t)64 << 20));  // 1.5 MB

  k_cvt_x<<<dim3(8192), 256, 0, stream>>>(x, Xb);
  k_cvt_w<<<dim3(1024, 1, 3), 256, 0, stream>>>(Wq, Wk, Wv, Wt);
  k_proj<<<dim3(128, 4, 3), 256, 0, stream>>>(Xb, Wt, bq, bk, bv, Qb, Kb, Vt);
  k_attn<<<dim3(128, 8), 256, 0, stream>>>(Qb, Kb, Vt, out);
}
